// VectorQuantizer_31791347925852
// MI455X (gfx1250) — matrix operand order repaired and checked by reading, not compiled, not run on hardware
//
#include <hip/hip_runtime.h>

// CDNA5 / gfx1250 VQ-VAE vector quantizer.
// Distances via split-precision f16 WMMA (hi+lo ~ f32 accuracy):
//   S = z . e^T computed as  zhi*ehi + zhi*elo + zlo*ehi  (f32 accumulate)
//   argmin_k ( ||e_k||^2 - 2 S )   (||z||^2 is row-constant, dropped)
// Outputs (concatenated float32): z_q (4194304), loss (1), indices (65536).

typedef __attribute__((ext_vector_type(16))) _Float16 v16h;
typedef __attribute__((ext_vector_type(8)))  _Float16 v8h;
typedef __attribute__((ext_vector_type(4)))  _Float16 v4h;
typedef __attribute__((ext_vector_type(8)))  float    v8f;
typedef __attribute__((ext_vector_type(4)))  float    v4f;

#define KCODES      1024
#define DDIM        64
#define ROWS_PER_WG 128
#define NUM_WG      512           // 65536 rows / 128
#define ESTRIDE     136           // halves per code row in LDS: 64 hi | 8 pad | 64 lo
#define ZQ_ELEMS    (64*64*32*32) // 4194304
#define LOSS_OFF    ((size_t)ZQ_ELEMS)
#define IDX_OFF     ((size_t)ZQ_ELEMS + 1)

union AF { v16h v; _Float16 e[16]; };
union BF { v16h v; v8h h[2]; };

__global__ __launch_bounds__(256) void vq_main(const float* __restrict__ Z,
                                               const float* __restrict__ E,
                                               float* __restrict__ out,
                                               float* __restrict__ ws)
{
  // 278528 + 32768 + 4096 + 512 + 1024 = 316928 B  (<= 320 KB/workgroup on CDNA5)
  __shared__ __align__(16) _Float16 Epk[KCODES * ESTRIDE];
  __shared__ __align__(16) float    Zs[ROWS_PER_WG * DDIM];
  __shared__ float e2s[KCODES];
  __shared__ int   idxs[ROWS_PER_WG];
  __shared__ float red[256];

  const int t = threadIdx.x;
  const int g = blockIdx.x;

  // ---- Stage embedding as f16 hi/lo pairs into LDS (coalesced 16B global loads) ----
  #pragma unroll 2
  for (int j = 0; j < 64; ++j) {
    int q    = t + j * 256;      // 0..16383
    int base = q * 4;            // element index into E (65536 total)
    int c = base >> 6, d = base & 63;
    v4f x = *(const v4f*)(E + base);
    v4h h, l;
    #pragma unroll
    for (int e = 0; e < 4; ++e) {
      _Float16 hh = (_Float16)x[e];
      h[e] = hh;
      l[e] = (_Float16)(x[e] - (float)hh);
    }
    *(v4h*)(Epk + c * ESTRIDE + d)      = h;   // hi run
    *(v4h*)(Epk + c * ESTRIDE + 72 + d) = l;   // lo run
  }

  // ---- ||e_k||^2 in f32 (rows are L2-hot after the pass above) ----
  {
    const v4f* E4 = (const v4f*)E;
    #pragma unroll
    for (int r = 0; r < 4; ++r) {
      int c = t * 4 + r;
      float s = 0.f;
      #pragma unroll
      for (int q = 0; q < 16; ++q) {
        v4f v = E4[c * 16 + q];
        s = fmaf(v[0], v[0], s);
        s = fmaf(v[1], v[1], s);
        s = fmaf(v[2], v[2], s);
        s = fmaf(v[3], v[3], s);
      }
      e2s[c] = s;
    }
  }

  // ---- Stage this workgroup's 128 z rows as exact f32 [row][d] ----
  const int    batch = g >> 3;
  const int    pos0  = (g & 7) << 7;
  const size_t zbase = (size_t)batch * 65536 + (size_t)pos0; // + d*1024 + i
  #pragma unroll 2
  for (int j = 0; j < 8; ++j) {
    int q  = t + j * 256;        // 0..2047
    int i4 = (q & 31) * 4;       // 4 consecutive spatial positions
    int d  = q >> 5;
    v4f zv = *(const v4f*)(Z + zbase + (size_t)d * 1024 + i4);
    #pragma unroll
    for (int e = 0; e < 4; ++e) Zs[(i4 + e) * DDIM + d] = zv[e];
  }
  __syncthreads();

  // ================= compute phase: one wave per 16-row strip =================
  const int lane = t & 31;
  const int wv   = t >> 5;      // wave id 0..7
  const int m16  = lane & 15;
  const int hf   = lane >> 4;   // lane half (wave32)
  const int kb   = hf * 8;

  // A fragments (hi/lo), built once from exact f32; contiguous K order matching B:
  //   half 0: K {0..15}, half 1: K {16..31} (+32 per chunk)
  AF ah[2], al[2];
  {
    const float* zr = &Zs[(wv * 16 + m16) * DDIM];
    #pragma unroll
    for (int q = 0; q < 2; ++q) {
      #pragma unroll
      for (int e = 0; e < 16; ++e) {
        int k   = 32 * q + 16 * hf + e;
        float x = zr[k];
        _Float16 hh = (_Float16)x;
        ah[q].e[e] = hh;
        al[q].e[e] = (_Float16)(x - (float)hh);
      }
    }
  }

  float best[8];
  int   bidx[8];
  #pragma unroll
  for (int v = 0; v < 8; ++v) { best[v] = 3.402823e38f; bidx[v] = 0; }

  #pragma unroll 2
  for (int tt = 0; tt < KCODES / 16; ++tt) {
    const int c = tt * 16 + m16;                 // this lane's code column
    const _Float16* er = &Epk[c * ESTRIDE];
    const int bo = 16 * hf;                      // B: lanes 0-15 hold K 0..15, 16-31 hold K 16..31
    BF bh0, bh1, bl0, bl1;
    bh0.h[0] = *(const v8h*)(er + bo);
    bh0.h[1] = *(const v8h*)(er + bo + 8);
    bh1.h[0] = *(const v8h*)(er + 32 + bo);
    bh1.h[1] = *(const v8h*)(er + 32 + bo + 8);
    bl0.h[0] = *(const v8h*)(er + 72 + bo);
    bl0.h[1] = *(const v8h*)(er + 72 + bo + 8);
    bl1.h[0] = *(const v8h*)(er + 72 + 32 + bo);
    bl1.h[1] = *(const v8h*)(er + 72 + 32 + bo + 8);
    const float e2c = e2s[c];

    v8f acc = {0.f, 0.f, 0.f, 0.f, 0.f, 0.f, 0.f, 0.f};
    // cross terms first, hi*hi last (f32 accumulate throughout)
    acc = __builtin_amdgcn_wmma_f32_16x16x32_f16(false, ah[0].v, false, bl0.v, (short)0, acc, false, false);
    acc = __builtin_amdgcn_wmma_f32_16x16x32_f16(false, al[0].v, false, bh0.v, (short)0, acc, false, false);
    acc = __builtin_amdgcn_wmma_f32_16x16x32_f16(false, ah[1].v, false, bl1.v, (short)0, acc, false, false);
    acc = __builtin_amdgcn_wmma_f32_16x16x32_f16(false, al[1].v, false, bh1.v, (short)0, acc, false, false);
    acc = __builtin_amdgcn_wmma_f32_16x16x32_f16(false, ah[0].v, false, bh0.v, (short)0, acc, false, false);
    acc = __builtin_amdgcn_wmma_f32_16x16x32_f16(false, ah[1].v, false, bh1.v, (short)0, acc, false, false);

    #pragma unroll
    for (int v = 0; v < 8; ++v) {
      float m = fmaf(-2.f, acc[v], e2c);         // ||e||^2 - 2 z.e
      if (m < best[v]) { best[v] = m; bidx[v] = c; }  // strict < keeps first (lowest idx)
    }
  }

  // reduce over the 16 lanes holding the same rows (slot v <-> row v + 8*hf)
  #pragma unroll
  for (int mask = 1; mask <= 8; mask <<= 1) {
    #pragma unroll
    for (int v = 0; v < 8; ++v) {
      float ov = __shfl_xor(best[v], mask, 32);
      int   oi = __shfl_xor(bidx[v], mask, 32);
      if (ov < best[v] || (ov == best[v] && oi < bidx[v])) { best[v] = ov; bidx[v] = oi; }
    }
  }
  if (m16 == 0) {
    #pragma unroll
    for (int v = 0; v < 8; ++v) idxs[wv * 16 + hf * 8 + v] = bidx[v];
  }
  __syncthreads();

  // ================= output phase: gather f32 rows, write z_q, loss =================
  float lsum = 0.f;
  #pragma unroll 4
  for (int j = 0; j < 32; ++j) {
    int idx = t + j * 256;
    int i = idx & 127, d = idx >> 7;
    int c = idxs[i];
    float ev = E[c * DDIM + d];                  // exact f32 embedding value
    size_t o = zbase + (size_t)d * 1024 + i;     // (B,D,H,W) layout, coalesced over i
    float diff = Z[o] - ev;
    out[o] = ev;                                 // forward value of z_q_st == z_q
    lsum = fmaf(diff, diff, lsum);
  }
  if (t < ROWS_PER_WG)
    out[IDX_OFF + (size_t)g * ROWS_PER_WG + t] = (float)idxs[t];

  red[t] = lsum;
  __syncthreads();
  #pragma unroll
  for (int s = 128; s > 0; s >>= 1) {            // fixed-order tree: deterministic
    if (t < s) red[t] += red[t + s];
    __syncthreads();
  }
  if (t == 0) ws[g] = red[0];
}

__global__ __launch_bounds__(512) void vq_finalize(const float* __restrict__ ws,
                                                   float* __restrict__ out)
{
  __shared__ float r[512];
  const int t = threadIdx.x;
  r[t] = ws[t];
  __syncthreads();
  #pragma unroll
  for (int s = 256; s > 0; s >>= 1) {
    if (t < s) r[t] += r[t + s];
    __syncthreads();
  }
  // loss = (1 + BETA) * mean((z - z_q)^2), BETA = 0.25, over 64*64*32*32 elems
  if (t == 0) out[LOSS_OFF] = r[0] * (1.25f / 4194304.f);
}

extern "C" void kernel_launch(void* const* d_in, const int* in_sizes, int n_in,
                              void* d_out, int out_size, void* d_ws, size_t ws_size,
                              hipStream_t stream) {
  const float* Z = (const float*)d_in[0];   // z_e: (64, 64, 32, 32) f32
  const float* E = (const float*)d_in[1];   // embedding: (1024, 64) f32
  float* out = (float*)d_out;               // [z_q | loss | indices] f32
  float* ws  = (float*)d_ws;                // 512 partial loss sums
  vq_main<<<NUM_WG, 256, 0, stream>>>(Z, E, out, ws);
  vq_finalize<<<1, 512, 0, stream>>>(ws, out);
}
